// GATlayer_89455578841370
// MI455X (gfx1250) — matrix operand order repaired and checked by reading, not compiled, not run on hardware
//
#include <hip/hip_runtime.h>

// GAT layer for MI455X (gfx1250, wave32).
//   Phase 1: H = X@W + b via V_WMMA_F32_16X16X4_F32 (full f32 precision for logits),
//            plus left/right attention projections and a transposed f16 copy of H.
//   Phase 2: flash-style fused leakyReLU/mask/online-softmax + att@H via
//            V_WMMA_F32_16X16X32_F16. adj (64 MB) is L2-resident (192 MB L2);
//            all adj/right traffic issued as 128-bit loads.

typedef __attribute__((ext_vector_type(2)))  float    v2f;
typedef __attribute__((ext_vector_type(4)))  float    v4f;
typedef __attribute__((ext_vector_type(4)))  int      v4i;
typedef __attribute__((ext_vector_type(8)))  float    v8f;
typedef __attribute__((ext_vector_type(16))) _Float16 v16h;

#define N_NODES 4096
#define CIN     128
#define COUT    64
#define BT      4
#define ALPHA_NEG 0.01f
#define MASK_NEG  (-1.0e12f)

// ---------------------------------------------------------------------------
// Phase 1: one wave (32 threads) per (bt, 16-row tile).
//   A fragment (f32 16x4): lane l holds row M=l&15; elems K = 2*(l>>4)+{0,1}
//   B fragment (f32 4x16): lane l holds col N=l&15; elems K = 2*(l>>4)+{0,1}
//   C/D (f32 16x16):       lane l holds col N=l&15; VGPR r -> row 8*(l>>4)+r
// ---------------------------------------------------------------------------
__global__ __launch_bounds__(32) void gat_phase1(
    const float* __restrict__ X, const float* __restrict__ W,
    const float* __restrict__ Wb, const float* __restrict__ a,
    float* __restrict__ left, float* __restrict__ right,
    _Float16* __restrict__ Ht)
{
    const int blk = blockIdx.x;          // 0 .. BT*256-1
    const int bt  = blk >> 8;
    const int n0  = (blk & 255) << 4;
    const int l   = threadIdx.x;
    const int col = l & 15;              // A-row index AND B/C column index
    const int hi  = l >> 4;              // which half of the wave

    __shared__ float Hs[16][COUT + 1];   // +1 pad: kill LDS bank conflicts

    v8f acc[4] = {};                     // 4 tiles of 16 columns -> Cout=64

    const float* Xrow = X + ((size_t)bt * N_NODES + (n0 + col)) * CIN;
    for (int k = 0; k < CIN; k += 4) {
        v2f Aa = *(const v2f*)(Xrow + k + 2 * hi);          // K = k+2hi, k+2hi+1
#pragma unroll
        for (int ct = 0; ct < 4; ++ct) {
            const float* wp = W + (size_t)(k + 2 * hi) * COUT + ct * 16 + col;
            v2f Bb;
            Bb.x = wp[0];          // W[k+2hi  ][c]
            Bb.y = wp[COUT];       // W[k+2hi+1][c]
            acc[ct] = __builtin_amdgcn_wmma_f32_16x16x4_f32(
                false, Aa, false, Bb, (short)0, acc[ct], false, false);
        }
    }

    // bias + park H tile in LDS (rows of this lane: 8*hi + r, column: col)
#pragma unroll
    for (int ct = 0; ct < 4; ++ct) {
        const int c = ct * 16 + col;
        const float bias = Wb[c];
#pragma unroll
        for (int r = 0; r < 8; ++r)
            Hs[8 * hi + r][c] = acc[ct][r] + bias;
    }
    __syncthreads();

    // left (lanes 0-15) / right (lanes 16-31): 64-wide dot with a
    {
        const float* av = a + hi * COUT;
        float s = 0.f;
#pragma unroll 8
        for (int c = 0; c < COUT; ++c) s += Hs[col][c] * av[c];
        float* dst = (hi == 0) ? left : right;
        dst[bt * N_NODES + n0 + col] = s;
    }

    // transposed f16 copy Ht[bt][c][n] for contiguous B-fragment loads in phase 2
#pragma unroll
    for (int e = 0; e < 32; ++e) {
        const int g  = l * 32 + e;       // 1024 elements / 32 lanes
        const int c  = g >> 4;
        const int rr = g & 15;
        Ht[((size_t)bt * COUT + c) * N_NODES + n0 + rr] = (_Float16)Hs[rr][c];
    }
}

// ---------------------------------------------------------------------------
// Phase 2: one wave per (bt, 16-row i-tile). Online softmax over j in chunks
// of 32, accumulating att@H with V_WMMA_F32_16X16X32_F16.
//   A fragment (f16 16x32): lane l row = l&15;
//       elem e -> K = (e&7) + 16*(e>>3) + 8*(l>>4)
//       => each lane's K-set is two contiguous 8-runs: [8hi,8hi+7], [16+8hi,16+8hi+7]
//       => adj / right are fetched with 128-bit vector loads.
//   B fragment (f16 32x16): lane l col = l&15; elem h -> K = (h&7) + 16*(h>>3) + 8*(l>>4) (same as A)
// ---------------------------------------------------------------------------
__global__ __launch_bounds__(32) void gat_phase2(
    const int* __restrict__ adj, const float* __restrict__ left,
    const float* __restrict__ right, const _Float16* __restrict__ Ht,
    const float* __restrict__ ab_p, float* __restrict__ out)
{
    const int blk = blockIdx.x;
    const int bt  = blk >> 8;
    const int i0  = (blk & 255) << 4;
    const int l   = threadIdx.x;
    const int row = l & 15;
    const int hi  = l >> 4;
    const float a_b = ab_p[0];

    __shared__ float f_lds[16];
    __shared__ float s_lds[16];

    const float  Lv      = left[bt * N_NODES + i0 + row];
    const int*   adj_row = adj + (size_t)(i0 + row) * N_NODES;   // adj shared across bt
    const float* right_b = right + bt * N_NODES;

    float m = -3.0e38f;   // running max
    float s = 0.0f;       // running denominator
    v8f acc[4] = {};

    for (int jc = 0; jc < N_NODES; jc += 32) {
        // two contiguous 8-element runs of j for this lane (32B-aligned)
        const int j0 = jc + (hi << 3);        // K = 8hi .. 8hi+7
        const int j1 = j0 + 16;               // K = 16+8hi .. 16+8hi+7

        v4i am[4];                            // adj, 16 values as 4 x b128
        am[0] = *(const v4i*)(adj_row + j0);
        am[1] = *(const v4i*)(adj_row + j0 + 4);
        am[2] = *(const v4i*)(adj_row + j1);
        am[3] = *(const v4i*)(adj_row + j1 + 4);
        v4f rv[4];                            // right, 16 values as 4 x b128
        rv[0] = *(const v4f*)(right_b + j0);
        rv[1] = *(const v4f*)(right_b + j0 + 4);
        rv[2] = *(const v4f*)(right_b + j1);
        rv[3] = *(const v4f*)(right_b + j1 + 4);

        // ---- logits for this lane's 16 A-fragment slots ----
        float ev[16];
        float mloc = -3.0e38f;
#pragma unroll
        for (int e = 0; e < 16; ++e) {        // e -> run (e>>2)&3? no: e/4 picks v4 block
            const int blkid = e >> 2;         // 0..3 over {run0 lo, run0 hi, run1 lo, run1 hi}
            const int sub   = e & 3;
            float x = Lv + rv[blkid][sub] + a_b;
            x = (x > 0.f) ? x : ALPHA_NEG * x;                 // LeakyReLU
            x = (am[blkid][sub] != 0) ? x : MASK_NEG;          // edge mask
            ev[e] = x;
            mloc = fmaxf(mloc, x);
        }
        // pair lanes l and l^16 hold the same row -> combine row stats
        mloc = fmaxf(mloc, __shfl_xor(mloc, 16, 32));
        const float new_m = fmaxf(m, mloc);
        const float f     = __expf(m - new_m);                 // accumulator rescale

        float csum = 0.f;
        v16h Ap;                               // ev[e] is already in A-fragment order
#pragma unroll
        for (int e = 0; e < 16; ++e) {
            const float p = __expf(ev[e] - new_m);
            csum += p;
            Ap[e] = (_Float16)p;               // weights in [0,1] -> f16
        }
        csum += __shfl_xor(csum, 16, 32);
        s = s * f + csum;
        m = new_m;

        // broadcast per-row rescale factor to the lanes owning those C rows
        if (hi == 0) f_lds[row] = f;
        __syncthreads();
        float fr[8];
#pragma unroll
        for (int r = 0; r < 8; ++r) fr[r] = f_lds[8 * hi + r];
        __syncthreads();

#pragma unroll
        for (int ct = 0; ct < 4; ++ct) {
#pragma unroll
            for (int r = 0; r < 8; ++r) acc[ct][r] *= fr[r];
            // B fragment: two 8-half runs (16B aligned) at K = 8hi.., 16+8hi.. from transposed H
            const _Float16* bp = Ht + ((size_t)bt * COUT + (ct * 16 + row)) * N_NODES
                                    + jc + (hi << 3);
            typedef __attribute__((ext_vector_type(8))) _Float16 v8h_t;
            const v8h_t B0 = *(const v8h_t*)bp;
            const v8h_t B1 = *(const v8h_t*)(bp + 16);
            const v16h Bm = __builtin_shufflevector(B0, B1, 0,1,2,3,4,5,6,7,8,9,10,11,12,13,14,15);
            acc[ct] = __builtin_amdgcn_wmma_f32_16x16x32_f16(
                false, Ap, false, Bm, (short)0, acc[ct], false, false);
        }
    }

    // ---- epilogue: divide by row sums, ReLU, store ----
    if (hi == 0) s_lds[row] = s;
    __syncthreads();
#pragma unroll
    for (int ct = 0; ct < 4; ++ct) {
        const int c = ct * 16 + row;
#pragma unroll
        for (int r = 0; r < 8; ++r) {
            const int rr = 8 * hi + r;
            float v = acc[ct][r] / s_lds[rr];
            v = fmaxf(v, 0.f);
            out[((size_t)bt * N_NODES + i0 + rr) * COUT + c] = v;
        }
    }
}

// ---------------------------------------------------------------------------
extern "C" void kernel_launch(void* const* d_in, const int* in_sizes, int n_in,
                              void* d_out, int out_size, void* d_ws, size_t ws_size,
                              hipStream_t stream) {
    const float* X   = (const float*)d_in[0];   // [2,2,4096,128]
    const int*   adj = (const int*)  d_in[1];   // [4096,4096]
    const float* W   = (const float*)d_in[2];   // [128,64]
    const float* Wb  = (const float*)d_in[3];   // [64]
    const float* a   = (const float*)d_in[4];   // [128]
    const float* ab  = (const float*)d_in[5];   // scalar
    float* out = (float*)d_out;                 // [2,2,4096,64]

    // workspace: left[4*4096] f32 | right[4*4096] f32 | Ht[4][64][4096] f16  (~2.25 MB)
    float*    left  = (float*)d_ws;
    float*    right = left + BT * N_NODES;
    _Float16* Ht    = (_Float16*)(right + BT * N_NODES);

    const dim3 grid(BT * (N_NODES / 16));
    gat_phase1<<<grid, 32, 0, stream>>>(X, W, Wb, a, left, right, Ht);
    gat_phase2<<<grid, 32, 0, stream>>>(adj, left, right, Ht, ab, out);
}
